// Decoder_10385230921896
// MI455X (gfx1250) — matrix operand order repaired and checked by reading, not compiled, not run on hardware
//
#include <hip/hip_runtime.h>

// ---------------------------------------------------------------------------
// Decoder (2-layer LSTM, teacher forcing) for MI455X / gfx1250, wave32 + WMMA.
//
//   1) W_ih/W_hh/W_out -> bf16 (f32 exponent range; WMMA f32 accumulate)
//   2) X = bf16(relu(emb[tok])) for all T*B rows (shifted tokens, SOS=1)
//   3) Xg = X @ W_ih0^T          (4096x2048x512 WMMA GEMM)
//   4) 128x lstm_step(layer0): gates = Xg[t] + h@W_hh0^T ; fused cell
//   5) Xg = Hl0seq @ W_ih1^T
//   6) 128x lstm_step(layer1) -> outs_bf
//   7) logits = outs @ W_out^T + b_out  (4096x32000x512), NT stores (524MB)
//   8) hT, cT tail
//
// CDNA5 specifics: v_wmma_f32_16x16x32_bf16, TDM tensor_load_to_lds for the
// A-panel / h_prev staging (s_wait_tensorcnt), fully-unrolled ping-pong k
// loops + sched_group_barrier pipelining (next iter's VMEM/DS loads grouped
// ahead of current iter's WMMAs), LDS ds_load_b128 A-tiles, NT output stores.
// ---------------------------------------------------------------------------

typedef __attribute__((ext_vector_type(16))) __bf16 v16bf;
typedef __attribute__((ext_vector_type(8)))  float  v8f;
typedef __attribute__((ext_vector_type(4)))  unsigned int v4u;
typedef __attribute__((ext_vector_type(8)))  int v8i;
typedef __attribute__((ext_vector_type(4)))  int v4i;
typedef unsigned short u16;

#define BSZ 32
#define TLEN 128
#define HDIM 512
#define NL 2
#define VOC 32000
#define GDIM 2048 /* 4*H */
#define KSTEPS (HDIM / 32) /* 16 */

#if __has_builtin(__builtin_amdgcn_tensor_load_to_lds)
#define HAVE_TDM 1
#else
#define HAVE_TDM 0
#endif

// sched_group_barrier masks: 0x020 VMEM read, 0x100 DS read, 0x008 MFMA/WMMA
#if __has_builtin(__builtin_amdgcn_sched_group_barrier)
#define SCHED_GROUP(mask, size) __builtin_amdgcn_sched_group_barrier(mask, size, 0)
#else
#define SCHED_GROUP(mask, size)
#endif

union BF16x16 { uint4 q[2]; v16bf v; };

__device__ __forceinline__ u16 f32_to_bf16(float f) {
  unsigned u = __float_as_uint(f);
  unsigned r = (u + 0x7FFFu + ((u >> 16) & 1u)) >> 16;  // round-nearest-even
  return (u16)r;
}

__device__ __forceinline__ v8f v8f_zero() {
  v8f z = {0.f, 0.f, 0.f, 0.f, 0.f, 0.f, 0.f, 0.f};
  return z;
}

// --------------------------- TDM 2D tile -> LDS ----------------------------
// D# per CDNA5 ISA. 2-byte elements, 2D tile rows x cols,
// row stride = stride_elems. Issued by one wave; wait with s_wait_tensorcnt.
// This toolchain's builtin takes 6 args: (g0 v4u, g1 v8i, v4i, v4i, v8i, cpol)
__device__ __forceinline__ void tdm_load_2d(unsigned lds_off, const void* gptr,
                                            unsigned rows, unsigned cols,
                                            unsigned stride_elems) {
#if HAVE_TDM
  unsigned long long ga = (unsigned long long)(size_t)gptr;
  v4u g0;
  g0[0] = 1u;                                   // count=1 (valid user D#)
  g0[1] = lds_off;                              // lds_addr (bytes)
  g0[2] = (unsigned)(ga & 0xFFFFFFFFu);         // global_addr lo
  g0[3] = (unsigned)((ga >> 32) & 0x01FFFFFFu)  // global_addr hi (57-bit)
          | (2u << 30);                         // type=2 ("image")
  v8i g1;
  g1[0] = (int)(1u << 16);                      // wg_mask=0, data_size=1 (2B)
  g1[1] = (int)((cols & 0xFFFFu) << 16);        // tensor_dim0[15:0]
  g1[2] = (int)(((cols >> 16) & 0xFFFFu) |      // tensor_dim0[31:16]
                ((rows & 0xFFFFu) << 16));      // tensor_dim1[15:0]
  g1[3] = (int)(((rows >> 16) & 0xFFFFu) |      // tensor_dim1[31:16]
                ((cols & 0xFFFFu) << 16));      // tile_dim0
  g1[4] = (int)(rows & 0xFFFFu);                // tile_dim1 (tile_dim2=0)
  g1[5] = (int)stride_elems;                    // tensor_dim0_stride[31:0]
  g1[6] = 0;                                    // stride hi, dim1_stride lo
  g1[7] = 0;
  v4i z4 = {0, 0, 0, 0};
  v8i z8 = {0, 0, 0, 0, 0, 0, 0, 0};
  __builtin_amdgcn_tensor_load_to_lds(g0, g1, z4, z4, z8, 0);
#endif
}

__device__ __forceinline__ void tdm_wait() {
#if HAVE_TDM
  __builtin_amdgcn_s_wait_tensorcnt(0);
#endif
}

// Cooperative fallback copy (used if TDM builtin is unavailable).
__device__ __forceinline__ void coop_copy_bf16(u16* dst_lds, const u16* src,
                                               int n_elems, int tid,
                                               int nthreads) {
  for (int i = tid * 8; i < n_elems; i += nthreads * 8)
    *(uint4*)(dst_lds + i) = *(const uint4*)(src + i);
}

// A tile (16x32 bf16, M x K): lane m = lane&15, h = lane>=16;
// elems 0..7 -> k0+8h+e, 8..15 -> k0+16+8h+(e-8). Two aligned 16B loads.
__device__ __forceinline__ v16bf lds_a_tile(const u16* sA, int row0, int k0,
                                            int lane) {
  int m = lane & 15, h = lane >> 4;
  const u16* p = sA + (row0 + m) * HDIM + k0 + 8 * h;
  BF16x16 r;
  r.q[0] = *(const uint4*)(p);
  r.q[1] = *(const uint4*)(p + 16);
  return r.v;
}

// B tile (32x16 bf16 = W^T tile from row-major W(N,K)): lane n = lane&15,
// elems 0..7 -> k0+8h+e, 8..15 -> k0+16+8h+(e-8). Two aligned 16B loads.
__device__ __forceinline__ v16bf load_b_tile(const u16* Bm, int col0, int k0,
                                             int lane) {
  int n = lane & 15, h = lane >> 4;
  const u16* p = Bm + (size_t)(col0 + n) * HDIM + k0 + 8 * h;
  BF16x16 r;
  r.q[0] = *(const uint4*)(p);
  r.q[1] = *(const uint4*)(p + 16);
  return r.v;
}

// ----------------------------- small kernels -------------------------------

__global__ void cvt_bf16_kernel(const float* __restrict__ src,
                                u16* __restrict__ dst, int n) {
  int i = blockIdx.x * blockDim.x + threadIdx.x;
  if (i < n) dst[i] = f32_to_bf16(src[i]);
}

__global__ void init_state_kernel(const float* __restrict__ h0,
                                  const float* __restrict__ c0,
                                  float* __restrict__ h_state,
                                  float* __restrict__ c_state,
                                  u16* __restrict__ hb0_l0,
                                  u16* __restrict__ hb0_l1) {
  int i = blockIdx.x * blockDim.x + threadIdx.x;  // over L*B*H
  if (i >= NL * BSZ * HDIM) return;
  h_state[i] = h0[i];
  c_state[i] = c0[i];
  u16 hb = f32_to_bf16(h0[i]);
  if (i < BSZ * HDIM) hb0_l0[i] = hb;
  else                hb0_l1[i - BSZ * HDIM] = hb;
}

__global__ void embed_relu_kernel(const int* __restrict__ target,
                                  const float* __restrict__ emb,
                                  u16* __restrict__ X) {
  int idx = blockIdx.x * blockDim.x + threadIdx.x;  // over T*B*H
  if (idx >= TLEN * BSZ * HDIM) return;
  int hcol = idx & (HDIM - 1);
  int r = idx >> 9;  // t*B + b
  int b = r & (BSZ - 1);
  int t = r >> 5;
  int tok = (t == 0) ? 1 : target[b * TLEN + (t - 1)];  // SOS = 1
  float x = emb[(size_t)tok * HDIM + hcol];
  x = x > 0.f ? x : 0.f;
  X[idx] = f32_to_bf16(x);
}

__global__ void finalize_kernel(const float* __restrict__ h_state,
                                const float* __restrict__ c_state,
                                float* __restrict__ out_tail) {
  int i = blockIdx.x * blockDim.x + threadIdx.x;  // over 2*L*B*H
  int n = NL * BSZ * HDIM;
  if (i < n)          out_tail[i] = h_state[i];
  else if (i < 2 * n) out_tail[i] = c_state[i - n];
}

// ------------------------- batch WMMA GEMM (plain) -------------------------
// C(M,N) f32 = A(M,K=512)bf16 @ B(N,K)bf16^T.  Block: 256 thr = 8 waves.
// WG: A-panel (32x512, 32KB) staged in LDS via TDM; wave computes 16x64.
// Fully unrolled k loop; sched_group_barrier pins the software pipeline:
// per k-iter (8 VMEM reads + 2 DS reads for iter i+1) then (4 WMMA of iter i).
__global__ __launch_bounds__(256) void gemm_bf16_kernel(
    const u16* __restrict__ A, const u16* __restrict__ Bm,
    float* __restrict__ C, int N) {
  __shared__ __align__(16) u16 sA[32 * HDIM];
  int lane = threadIdx.x & 31, wave = threadIdx.x >> 5;
#if HAVE_TDM
  if (wave == 0)
    tdm_load_2d((unsigned)(size_t)sA, A + (size_t)blockIdx.y * 32 * HDIM, 32,
                HDIM, HDIM);
  tdm_wait();
#else
  coop_copy_bf16(sA, A + (size_t)blockIdx.y * 32 * HDIM, 32 * HDIM,
                 threadIdx.x, 256);
#endif
  __syncthreads();

  int aRow0 = (wave & 1) * 16;
  int nb = blockIdx.x * 256 + (wave >> 1) * 64;
  v8f acc[4] = {v8f_zero(), v8f_zero(), v8f_zero(), v8f_zero()};

  v16bf a[2], b[2][4];
  a[0] = lds_a_tile(sA, aRow0, 0, lane);
#pragma unroll
  for (int j = 0; j < 4; ++j) b[0][j] = load_b_tile(Bm, nb + j * 16, 0, lane);

#pragma unroll
  for (int it = 0; it < KSTEPS; ++it) {
    const int cur = it & 1, nx = cur ^ 1;
    if (it < KSTEPS - 1) {
      a[nx] = lds_a_tile(sA, aRow0, (it + 1) * 32, lane);
#pragma unroll
      for (int j = 0; j < 4; ++j)
        b[nx][j] = load_b_tile(Bm, nb + j * 16, (it + 1) * 32, lane);
    }
#pragma unroll
    for (int j = 0; j < 4; ++j)
      acc[j] = __builtin_amdgcn_wmma_f32_16x16x32_bf16(
          false, a[cur], false, b[cur][j], (short)0, acc[j], false, false);
    SCHED_GROUP(0x020, 8);  // 8 VMEM reads (next iter's B tiles)
    SCHED_GROUP(0x100, 2);  // 2 DS reads (next iter's A tile)
    SCHED_GROUP(0x008, 4);  // 4 WMMA (this iter)
  }

  int n = lane & 15, h = lane >> 4;
  int m0 = blockIdx.y * 32 + aRow0;
#pragma unroll
  for (int j = 0; j < 4; ++j)
#pragma unroll
    for (int r = 0; r < 8; ++r)
      C[(size_t)(m0 + r + 8 * h) * N + nb + j * 16 + n] = acc[j][r];
}

// -------------------- logits GEMM: permuted NT store + bias ----------------
__global__ __launch_bounds__(256) void gemm_logits_kernel(
    const u16* __restrict__ A, const u16* __restrict__ Bm,
    const float* __restrict__ bias, float* __restrict__ out) {
  __shared__ __align__(16) u16 sA[32 * HDIM];
  int lane = threadIdx.x & 31, wave = threadIdx.x >> 5;
#if HAVE_TDM
  if (wave == 0)
    tdm_load_2d((unsigned)(size_t)sA, A + (size_t)blockIdx.y * 32 * HDIM, 32,
                HDIM, HDIM);
  tdm_wait();
#else
  coop_copy_bf16(sA, A + (size_t)blockIdx.y * 32 * HDIM, 32 * HDIM,
                 threadIdx.x, 256);
#endif
  __syncthreads();

  int aRow0 = (wave & 1) * 16;
  int nb = blockIdx.x * 256 + (wave >> 1) * 64;
  v8f acc[4] = {v8f_zero(), v8f_zero(), v8f_zero(), v8f_zero()};

  v16bf a[2], b[2][4];
  a[0] = lds_a_tile(sA, aRow0, 0, lane);
#pragma unroll
  for (int j = 0; j < 4; ++j) b[0][j] = load_b_tile(Bm, nb + j * 16, 0, lane);

#pragma unroll
  for (int it = 0; it < KSTEPS; ++it) {
    const int cur = it & 1, nx = cur ^ 1;
    if (it < KSTEPS - 1) {
      a[nx] = lds_a_tile(sA, aRow0, (it + 1) * 32, lane);
#pragma unroll
      for (int j = 0; j < 4; ++j)
        b[nx][j] = load_b_tile(Bm, nb + j * 16, (it + 1) * 32, lane);
    }
#pragma unroll
    for (int j = 0; j < 4; ++j)
      acc[j] = __builtin_amdgcn_wmma_f32_16x16x32_bf16(
          false, a[cur], false, b[cur][j], (short)0, acc[j], false, false);
    SCHED_GROUP(0x020, 8);  // 8 VMEM reads (next iter's B tiles)
    SCHED_GROUP(0x100, 2);  // 2 DS reads (next iter's A tile)
    SCHED_GROUP(0x008, 4);  // 4 WMMA (this iter)
  }

  int n = lane & 15, h = lane >> 4;
  int m0 = blockIdx.y * 32 + aRow0;
#pragma unroll
  for (int j = 0; j < 4; ++j) {
    int col = nb + j * 16 + n;
    float bv = bias[col];
#pragma unroll
    for (int r = 0; r < 8; ++r) {
      int rg = m0 + r + 8 * h;  // t*B + b
      int t = rg >> 5, b2 = rg & (BSZ - 1);
      // streaming (non-temporal) store: don't evict L2-resident weights
      __builtin_nontemporal_store(
          acc[j][r] + bv, &out[((size_t)(b2 * TLEN + t)) * VOC + col]);
    }
  }
}

// ----------------------------- LSTM step -----------------------------------
// One launch per (t, layer). grid = 32 WGs (16 h-cols each), block = 256 (8
// waves = 2 M-tiles x 4 gates). gates = xg_t + h_prev @ W_hh^T; h_prev staged
// in LDS via TDM; gate tiles staged in LDS; fused elementwise cell update.
__global__ __launch_bounds__(256) void lstm_step_kernel(
    const float* __restrict__ xg_t,    // (B, 4H) precomputed input gates
    const u16*   __restrict__ whh,     // (4H, H) bf16
    const float* __restrict__ b_ih,    // (4H)
    const float* __restrict__ b_hh,    // (4H)
    const u16*   __restrict__ h_in,    // (B, H) bf16 (read buffer)
    u16*         __restrict__ h_out,   // (B, H) bf16 (write buffer)
    float*       __restrict__ h_state, // (B, H) f32
    float*       __restrict__ c_state, // (B, H) f32
    u16*         __restrict__ seq_out) // (B, H) bf16 slice at time t
{
  __shared__ __align__(16) u16 sH[BSZ * HDIM];   // 32KB staged h_prev
  __shared__ float gsm[4][BSZ][16];              // gate tiles
  int lane = threadIdx.x & 31, wave = threadIdx.x >> 5;
#if HAVE_TDM
  if (wave == 0)
    tdm_load_2d((unsigned)(size_t)sH, h_in, BSZ, HDIM, HDIM);
  tdm_wait();
#else
  coop_copy_bf16(sH, h_in, BSZ * HDIM, threadIdx.x, 256);
#endif
  __syncthreads();

  int gate = wave & 3, m0 = (wave >> 2) * 16;
  int j0 = blockIdx.x * 16;
  int col0 = gate * HDIM + j0;
  int n = lane & 15, h = lane >> 4;

  v8f acc;
#pragma unroll
  for (int r = 0; r < 8; ++r)
    acc[r] = xg_t[(size_t)(m0 + r + 8 * h) * GDIM + col0 + n];

  v16bf a[2], b[2];
  a[0] = lds_a_tile(sH, m0, 0, lane);
  b[0] = load_b_tile(whh, col0, 0, lane);
#pragma unroll
  for (int it = 0; it < KSTEPS; ++it) {
    const int cur = it & 1, nx = cur ^ 1;
    if (it < KSTEPS - 1) {
      a[nx] = lds_a_tile(sH, m0, (it + 1) * 32, lane);
      b[nx] = load_b_tile(whh, col0, (it + 1) * 32, lane);
    }
    acc = __builtin_amdgcn_wmma_f32_16x16x32_bf16(false, a[cur], false, b[cur],
                                                  (short)0, acc, false, false);
    SCHED_GROUP(0x020, 2);  // 2 VMEM reads (next iter's B tile)
    SCHED_GROUP(0x100, 2);  // 2 DS reads (next iter's A tile)
    SCHED_GROUP(0x008, 1);  // 1 WMMA (this iter)
  }

#pragma unroll
  for (int r = 0; r < 8; ++r) gsm[gate][m0 + r + 8 * h][n] = acc[r];
  __syncthreads();

  // 32 rows x 16 cols = 512 elements; 2 per thread.
  int tid = threadIdx.x;
#pragma unroll
  for (int e2 = 0; e2 < 2; ++e2) {
    int e = tid * 2 + e2;
    int br = e >> 4;  // batch row
    int jj = e & 15;
    int j = j0 + jj;
    float iv = gsm[0][br][jj] + b_ih[j]            + b_hh[j];
    float fv = gsm[1][br][jj] + b_ih[HDIM + j]     + b_hh[HDIM + j];
    float gv = gsm[2][br][jj] + b_ih[2 * HDIM + j] + b_hh[2 * HDIM + j];
    float ov = gsm[3][br][jj] + b_ih[3 * HDIM + j] + b_hh[3 * HDIM + j];
    float c_old = c_state[br * HDIM + j];
    float si = 1.f / (1.f + __expf(-iv));
    float sf = 1.f / (1.f + __expf(-fv));
    float so = 1.f / (1.f + __expf(-ov));
    float tg = tanhf(gv);
    float cn = sf * c_old + si * tg;
    float hn = so * tanhf(cn);
    c_state[br * HDIM + j] = cn;
    h_state[br * HDIM + j] = hn;
    u16 hb = f32_to_bf16(hn);
    h_out[br * HDIM + j] = hb;
    seq_out[br * HDIM + j] = hb;
  }
}

// ------------------------------- host side ---------------------------------

extern "C" void kernel_launch(void* const* d_in, const int* in_sizes, int n_in,
                              void* d_out, int out_size, void* d_ws,
                              size_t ws_size, hipStream_t stream) {
  (void)in_sizes; (void)n_in; (void)out_size; (void)ws_size;
  const float* enc    = (const float*)d_in[0]; (void)enc;  // unused by ref
  const float* h0     = (const float*)d_in[1];
  const float* c0     = (const float*)d_in[2];
  const int*   target = (const int*)d_in[3];
  const float* emb    = (const float*)d_in[4];
  const float* W_ih   = (const float*)d_in[5];
  const float* W_hh   = (const float*)d_in[6];
  const float* b_ih   = (const float*)d_in[7];
  const float* b_hh   = (const float*)d_in[8];
  const float* W_out  = (const float*)d_in[9];
  const float* b_out  = (const float*)d_in[10];
  float* out = (float*)d_out;

  char* ws = (char*)d_ws;
  size_t off = 0;
  auto carve = [&](size_t bytes) {
    void* p = ws + off;
    off += (bytes + 255) & ~(size_t)255;
    return p;
  };
  const size_t BH  = (size_t)BSZ * HDIM;
  const size_t WG1 = (size_t)GDIM * HDIM;
  u16*   Wihb  = (u16*)carve(NL * WG1 * 2);
  u16*   Whhb  = (u16*)carve(NL * WG1 * 2);
  u16*   Woutb = (u16*)carve((size_t)VOC * HDIM * 2);
  u16*   Xb    = (u16*)carve((size_t)TLEN * BH * 2);
  float* Xg    = (float*)carve((size_t)TLEN * BSZ * GDIM * 4);  // reused
  u16*   Hl0   = (u16*)carve((size_t)TLEN * BH * 2);
  u16*   Outs  = (u16*)carve((size_t)TLEN * BH * 2);
  float* h_st  = (float*)carve(NL * BH * 4);
  float* c_st  = (float*)carve(NL * BH * 4);
  u16*   hb_l0 = (u16*)carve(2 * BH * 2);
  u16*   hb_l1 = (u16*)carve(2 * BH * 2);

  {
    int nW = (int)(NL * WG1);
    cvt_bf16_kernel<<<(nW + 255) / 256, 256, 0, stream>>>(W_ih, Wihb, nW);
    cvt_bf16_kernel<<<(nW + 255) / 256, 256, 0, stream>>>(W_hh, Whhb, nW);
    int nWo = VOC * HDIM;
    cvt_bf16_kernel<<<(nWo + 255) / 256, 256, 0, stream>>>(W_out, Woutb, nWo);
  }
  init_state_kernel<<<(NL * BSZ * HDIM + 255) / 256, 256, 0, stream>>>(
      h0, c0, h_st, c_st, hb_l0, hb_l1);
  embed_relu_kernel<<<(TLEN * BSZ * HDIM + 255) / 256, 256, 0, stream>>>(
      target, emb, Xb);

  // Xg = X @ W_ih0^T
  gemm_bf16_kernel<<<dim3(GDIM / 256, (TLEN * BSZ) / 32), 256, 0, stream>>>(
      Xb, Wihb, Xg, GDIM);

  for (int t = 0; t < TLEN; ++t) {
    lstm_step_kernel<<<HDIM / 16, 256, 0, stream>>>(
        Xg + (size_t)t * BSZ * GDIM, Whhb, b_ih, b_hh,
        hb_l0 + (size_t)(t & 1) * BH, hb_l0 + (size_t)((t + 1) & 1) * BH,
        h_st, c_st, Hl0 + (size_t)t * BH);
  }

  // Xg = Hl0seq @ W_ih1^T
  gemm_bf16_kernel<<<dim3(GDIM / 256, (TLEN * BSZ) / 32), 256, 0, stream>>>(
      Hl0, Wihb + WG1, Xg, GDIM);

  for (int t = 0; t < TLEN; ++t) {
    lstm_step_kernel<<<HDIM / 16, 256, 0, stream>>>(
        Xg + (size_t)t * BSZ * GDIM, Whhb + WG1, b_ih + GDIM, b_hh + GDIM,
        hb_l1 + (size_t)(t & 1) * BH, hb_l1 + (size_t)((t + 1) & 1) * BH,
        h_st + BH, c_st + BH, Outs + (size_t)t * BH);
  }

  // logits -> d_out (B,T,V) f32
  gemm_logits_kernel<<<dim3(VOC / 256, (TLEN * BSZ) / 32), 256, 0, stream>>>(
      Outs, Woutb, b_out, out);

  finalize_kernel<<<(2 * NL * BSZ * HDIM + 255) / 256, 256, 0, stream>>>(
      h_st, c_st, out + (size_t)BSZ * TLEN * VOC);
}
